// RoPEVariantLayer_71210557768293
// MI455X (gfx1250) — matrix operand order repaired and checked by reading, not compiled, not run on hardware
//
#include <hip/hip_runtime.h>

typedef __attribute__((ext_vector_type(16))) _Float16 v16h;
typedef __attribute__((ext_vector_type(8)))  float    v8f;
typedef __attribute__((ext_vector_type(4)))  float    v4f;

#define T_SEQ   4096
#define H_HEADS 16
#define D_DIM   64
#define QTILE   16                 // query rows per wave
#define WAVES   8                  // waves per block (wave32)
#define BLOCK_Q (QTILE * WAVES)    // 128 query rows per block
#define KTILE   64                 // keys per inner tile (2x WMMA K=32)

// Single-instruction DPP butterfly steps: x = op(dpp(x), x). Stays within a
// 16-lane row. quad_perm xor1/xor2, then half-mirror (xor4 after quad
// uniformity), then mirror (xor8 after half uniformity).
#define DPP_MAXS(x, CTRL) \
    asm("v_max_num_f32_dpp %0, %1, %2 " CTRL \
        " row_mask:0xf bank_mask:0xf bound_ctrl:1" \
        : "=v"(x) : "v"(x), "v"(x))
#define DPP_ADDS(x, CTRL) \
    asm("v_add_f32_dpp %0, %1, %2 " CTRL \
        " row_mask:0xf bank_mask:0xf bound_ctrl:1" \
        : "=v"(x) : "v"(x), "v"(x))

#define RED16_MAX(x) do { \
    DPP_MAXS(x, "quad_perm:[1,0,3,2]"); \
    DPP_MAXS(x, "quad_perm:[2,3,0,1]"); \
    DPP_MAXS(x, "row_half_mirror");     \
    DPP_MAXS(x, "row_mirror"); } while (0)

#define RED16_SUM(x) do { \
    DPP_ADDS(x, "quad_perm:[1,0,3,2]"); \
    DPP_ADDS(x, "quad_perm:[2,3,0,1]"); \
    DPP_ADDS(x, "row_half_mirror");     \
    DPP_ADDS(x, "row_mirror"); } while (0)

static __device__ __forceinline__ unsigned pkh2(float a, float b) {
    // v_cvt_pk_rtz_f16_f32 -> packed half2 as u32
    return __builtin_bit_cast(unsigned, __builtin_amdgcn_cvt_pkrtz(a, b));
}

// Issue this thread's share of one 64x64 K/V tile as async global->LDS b128
// copies (ASYNCcnt-tracked DMA, no VGPR data movement, no LOADcnt).
static __device__ __forceinline__ void async_stage_tile(
        const float* __restrict__ gK, const float* __restrict__ gV,
        float (*bK)[D_DIM], float (*bV)[D_DIM], int tid)
{
    #pragma unroll
    for (int i = 0; i < 4; ++i) {
        const int idx = tid + i * 256;       // 1024 float4 per 64x64 tile
        const int key = idx >> 4;
        const int d4  = (idx & 15) * 4;
        const float* gk = gK + (size_t)key * D_DIM + d4;
        const float* gv = gV + (size_t)key * D_DIM + d4;
        const unsigned lk = (unsigned)(unsigned long long)&bK[key][d4];
        const unsigned lv = (unsigned)(unsigned long long)&bV[key][d4];
        asm volatile("global_load_async_to_lds_b128 %0, %1, off"
                     :: "v"(lk), "v"(gk) : "memory");
        asm volatile("global_load_async_to_lds_b128 %0, %1, off"
                     :: "v"(lv), "v"(gv) : "memory");
    }
}

__global__ __launch_bounds__(256)
void alibi_flash_attn_kernel(const float* __restrict__ Q,
                             const float* __restrict__ K,
                             const float* __restrict__ V,
                             float* __restrict__ O)
{
    // Double-buffered f32 staging tiles (async DMA targets) + f16 operand tiles.
    __shared__ __align__(16) float    stgK[2][KTILE][D_DIM];   // 32 KB
    __shared__ __align__(16) float    stgV[2][KTILE][D_DIM];   // 32 KB
    __shared__ __align__(16) _Float16 sK[KTILE][D_DIM];        // 8 KB  [key][d]
    __shared__ __align__(16) _Float16 sVt[D_DIM][KTILE];       // 8 KB  [d][key]
    __shared__ __align__(16) _Float16 sP[WAVES][QTILE][KTILE]; // 16 KB P transpose

    const int tid   = threadIdx.x;
    const int wave  = tid >> 5;
    const int lane  = tid & 31;
    const int lhalf = lane >> 4;
    const int l16   = lane & 15;

    const int h     = blockIdx.y;
    const int qbase = blockIdx.x * BLOCK_Q + wave * QTILE;

    const float LOG2E = 1.4426950408889634f;
    const float slope = exp2f(-8.0f * (float)(h + 1) / (float)H_HEADS);
    const float sl2   = 0.125f * LOG2E;   // (1/sqrt(64)) * log2(e)
    const float bl2   = slope  * LOG2E;   // ALiBi slope in log2 domain

    const float* Qh = Q + (size_t)h * T_SEQ * D_DIM;
    const float* Kh = K + (size_t)h * T_SEQ * D_DIM;
    const float* Vh = V + (size_t)h * T_SEQ * D_DIM;
    float*       Oh = O + (size_t)h * T_SEQ * D_DIM;

    // ---- Q A-fragments (16x32 f16 each; 2 fragments cover D=64) ----
    v16h qa[2];
    {
        const float* qp = Qh + (size_t)(qbase + l16) * D_DIM;
        #pragma unroll
        for (int f = 0; f < 2; ++f) {
            const int dbase = 32 * f + 16 * lhalf;
            #pragma unroll
            for (int e = 0; e < 8; ++e) {
                qa[f][e]     = (_Float16)qp[dbase + e];
                qa[f][e + 8] = (_Float16)qp[dbase + 8 + e];
            }
        }
    }

    // Float query positions per C-fragment row (row r+8*lhalf), running key pos.
    float qf[8];
    #pragma unroll
    for (int r = 0; r < 8; ++r) qf[r] = (float)(qbase + r + 8 * lhalf);
    float kf = (float)l16;

    // Online-softmax state kept in log2 units.
    float m_i[8], l_i[8];
    #pragma unroll
    for (int r = 0; r < 8; ++r) { m_i[r] = -1e30f; l_i[r] = 0.0f; }

    v8f oacc[4];
    #pragma unroll
    for (int c = 0; c < 4; ++c) oacc[c] = (v8f){};

    // Prime the async pipeline with tile 0.
    async_stage_tile(Kh, Vh, stgK[0], stgV[0], tid);

    for (int kt = 0; kt < T_SEQ; kt += KTILE) {
        const int buf = (kt >> 6) & 1;

        // Prefetch next tile behind this tile's compute, then wait for the
        // OLDER batch only (async loads complete in order): <=8 in flight.
        if (kt + KTILE < T_SEQ) {
            async_stage_tile(Kh + (size_t)(kt + KTILE) * D_DIM,
                             Vh + (size_t)(kt + KTILE) * D_DIM,
                             stgK[buf ^ 1], stgV[buf ^ 1], tid);
            asm volatile("s_wait_asynccnt 8" ::: "memory");
        } else {
            asm volatile("s_wait_asynccnt 0" ::: "memory");
        }
        __syncthreads();   // staged f32 tile visible; previous sK/sVt readers done

        // ---- cooperative convert: stg f32 -> packed f16 sK / transposed sVt ----
        {
            const int kq = tid >> 4;          // keys 4kq..4kq+3
            const int d4 = (tid & 15) * 4;
            v4f kr[4], vr[4];
            #pragma unroll
            for (int j = 0; j < 4; ++j) {
                kr[j] = *(const v4f*)&stgK[buf][4 * kq + j][d4];
                vr[j] = *(const v4f*)&stgV[buf][4 * kq + j][d4];
            }
            #pragma unroll
            for (int j = 0; j < 4; ++j) {
                unsigned* pk = (unsigned*)&sK[4 * kq + j][d4];
                pk[0] = pkh2(kr[j][0], kr[j][1]);
                pk[1] = pkh2(kr[j][2], kr[j][3]);
                unsigned* pv = (unsigned*)&sVt[d4 + j][4 * kq];
                pv[0] = pkh2(vr[0][j], vr[1][j]);
                pv[1] = pkh2(vr[2][j], vr[3][j]);
            }
        }
        __syncthreads();

        // ---- scores: four 16x16 tiles, contraction over D via 2 WMMAs each ----
        v8f sc[4];
        #pragma unroll
        for (int t = 0; t < 4; ++t) {
            v8f cacc = (v8f){};
            #pragma unroll
            for (int f = 0; f < 2; ++f) {
                v16h kb;
                const _Float16* src = &sK[16 * t + l16][32 * f + 16 * lhalf];
                #pragma unroll
                for (int e = 0; e < 16; ++e) kb[e] = src[e];
                cacc = __builtin_amdgcn_wmma_f32_16x16x32_f16(
                           false, qa[f], false, kb, (short)0, cacc, false, false);
            }
            sc[t] = cacc;
        }

        // ---- ALiBi bias + online softmax (log2 domain); DPP reductions ----
        float mnew[8];
        #pragma unroll
        for (int r = 0; r < 8; ++r) {
            const float d0 = qf[r] - kf;
            const float d1 = d0 - 16.0f;
            const float d2 = d0 - 32.0f;
            const float d3 = d0 - 48.0f;
            float s0 = fmaf(sc[0][r], sl2, -bl2 * fabsf(d0));
            float s1 = fmaf(sc[1][r], sl2, -bl2 * fabsf(d1));
            float s2 = fmaf(sc[2][r], sl2, -bl2 * fabsf(d2));
            float s3 = fmaf(sc[3][r], sl2, -bl2 * fabsf(d3));
            sc[0][r] = s0; sc[1][r] = s1; sc[2][r] = s2; sc[3][r] = s3;
            float mx = fmaxf(fmaxf(s0, s1), fmaxf(s2, s3));
            RED16_MAX(mx);
            mnew[r] = fmaxf(m_i[r], mx);
        }
        #pragma unroll
        for (int r = 0; r < 8; ++r) {
            const float mn = mnew[r];
            float p0 = exp2f(sc[0][r] - mn);
            float p1 = exp2f(sc[1][r] - mn);
            float p2 = exp2f(sc[2][r] - mn);
            float p3 = exp2f(sc[3][r] - mn);
            sc[0][r] = p0; sc[1][r] = p1; sc[2][r] = p2; sc[3][r] = p3;
            float rs = (p0 + p1) + (p2 + p3);
            RED16_SUM(rs);
            const float corr = exp2f(m_i[r] - mn);
            l_i[r] = fmaf(l_i[r], corr, rs);
            m_i[r] = mn;
            #pragma unroll
            for (int c = 0; c < 4; ++c) oacc[c][r] *= corr;
        }
        kf += (float)KTILE;

        // ---- transpose P (C-layout regs -> LDS -> A-layout regs), wave-private ----
        #pragma unroll
        for (int t = 0; t < 4; ++t)
            #pragma unroll
            for (int r = 0; r < 8; ++r)
                sP[wave][r + 8 * lhalf][16 * t + l16] = (_Float16)sc[t][r];
        asm volatile("s_wait_dscnt 0" ::: "memory");

        v16h pa[2];
        #pragma unroll
        for (int g = 0; g < 2; ++g) {
            const _Float16* src = &sP[wave][l16][32 * g];
            #pragma unroll
            for (int e = 0; e < 8; ++e) {
                pa[g][e]     = src[16 * lhalf + e];
                pa[g][e + 8] = src[16 * lhalf + 8 + e];
            }
        }

        // ---- O += P x V : contiguous B-fragments from sVt, K=64 in 2 steps ----
        #pragma unroll
        for (int c = 0; c < 4; ++c) {
            #pragma unroll
            for (int g = 0; g < 2; ++g) {
                v16h vb;
                const _Float16* src = &sVt[16 * c + l16][32 * g + 16 * lhalf];
                #pragma unroll
                for (int e = 0; e < 16; ++e) vb[e] = src[e];
                oacc[c] = __builtin_amdgcn_wmma_f32_16x16x32_f16(
                              false, pa[g], false, vb, (short)0, oacc[c],
                              false, false);
            }
        }
    }

    // ---- finalize: O /= l, write back ----
    #pragma unroll
    for (int r = 0; r < 8; ++r) {
        const float inv  = 1.0f / l_i[r];
        const int   qrow = qbase + r + 8 * lhalf;
        float* op = Oh + (size_t)qrow * D_DIM;
        #pragma unroll
        for (int c = 0; c < 4; ++c)
            op[16 * c + l16] = oacc[c][r] * inv;
    }
}

extern "C" void kernel_launch(void* const* d_in, const int* in_sizes, int n_in,
                              void* d_out, int out_size, void* d_ws, size_t ws_size,
                              hipStream_t stream) {
    const float* Q = (const float*)d_in[0];
    const float* K = (const float*)d_in[1];
    const float* V = (const float*)d_in[2];
    float*       O = (float*)d_out;

    dim3 grid(T_SEQ / BLOCK_Q, H_HEADS, 1);   // (32, 16)
    dim3 block(256, 1, 1);                    // 8 waves (wave32)
    alibi_flash_attn_kernel<<<grid, block, 0, stream>>>(Q, K, V, O);
}
